// GCNEncoder_Commoninput_5858335392189
// MI455X (gfx1250) — matrix operand order repaired and checked by reading, not compiled, not run on hardware
//
#include <hip/hip_runtime.h>
#include <math.h>

// ---------------------------------------------------------------------------
// Model dims (fixed by reference)
// ---------------------------------------------------------------------------
#define NN 50000
#define EE 800000
// conv2: IN=256 -> 512 (Q|K|V|S, 2 heads x 64)
// conv3: 128    -> 256 (Q|K|V|S, 1 head)
// conv4: 256    -> 256 (Q|K|V|S, 1 head)

typedef __attribute__((ext_vector_type(16))) _Float16 v16h;
typedef __attribute__((ext_vector_type(8)))  _Float16 v8h;
typedef __attribute__((ext_vector_type(8)))  float    v8f;

// ---------------------------------------------------------------------------
// Utility kernels
// ---------------------------------------------------------------------------
__global__ __launch_bounds__(256) void fill_f32(float* __restrict__ p, float v, size_t cnt) {
  size_t t = (size_t)blockIdx.x * blockDim.x + threadIdx.x;
  if (t < cnt) p[t] = v;
}

__global__ __launch_bounds__(256) void cast_f32_to_f16(const float* __restrict__ src,
                                                       _Float16* __restrict__ dst, size_t cnt) {
  size_t t = (size_t)blockIdx.x * blockDim.x + threadIdx.x;
  if (t < cnt) dst[t] = (_Float16)src[t];
}

// Fuse (Wq|Wk|Wv|Ws) [K x Cpart each] into transposed f16 weight Wt[NC][K],
// NC = 4*Cpart, and fused f32 bias[NC].
__global__ __launch_bounds__(256) void prep_w(const float* __restrict__ Wq, const float* __restrict__ Wk,
                                              const float* __restrict__ Wv, const float* __restrict__ Ws,
                                              const float* __restrict__ bq, const float* __restrict__ bk,
                                              const float* __restrict__ bv, const float* __restrict__ bs,
                                              _Float16* __restrict__ Wt, float* __restrict__ bfused,
                                              int K, int Cpart) {
  int t = blockIdx.x * blockDim.x + threadIdx.x;
  int NC = 4 * Cpart;
  if (t >= NC * K) return;
  int n = t / K, k = t % K;
  int part = n / Cpart, nc = n % Cpart;
  const float* W = (part == 0) ? Wq : (part == 1) ? Wk : (part == 2) ? Wv : Ws;
  Wt[t] = (_Float16)W[(size_t)k * Cpart + nc];
  if (k == 0) {
    const float* b = (part == 0) ? bq : (part == 1) ? bk : (part == 2) ? bv : bs;
    bfused[n] = b[nc];
  }
}

// ---------------------------------------------------------------------------
// WMMA GEMM: Y[M x NC] = Xh[M x K] * Wt^T + bias  (Wt stored [NC][K], f16)
// One wave per 16x64 output tile (A fragment reused across 4 WMMA calls).
// Fragment layouts per CDNA5 ISA 7.12.2 (f16 16x16x32).
// ---------------------------------------------------------------------------
__global__ __launch_bounds__(256) void wmma_gemm_f16(const _Float16* __restrict__ Xh,
                                                     const _Float16* __restrict__ Wt,
                                                     const float* __restrict__ bias,
                                                     float* __restrict__ Y,
                                                     int M, int K, int NC) {
  const int wave = threadIdx.x >> 5;
  const int lane = threadIdx.x & 31;
  const int tiles_n = NC >> 6;
  const int tile = blockIdx.x * (blockDim.x >> 5) + wave;
  const int total = (M >> 4) * tiles_n;
  if (tile >= total) return;                 // wave-uniform: EXEC stays all-ones for WMMA
  const int tm = tile / tiles_n;
  const int tn = tile % tiles_n;
  const int m0 = tm << 4;
  const int n0 = tn << 6;
  const int lm = lane & 15;                  // lane's M (A) / N (B,C)
  const int lh = lane >> 4;                  // lane half

  v8f acc[4] = {};

  // A: lane holds row m0+lm; halves j=0..7 -> K=kbase+j, j=8..15 -> K=kbase+16+(j-8)
  const _Float16* arow = Xh + (size_t)(m0 + lm) * K + (lh << 3);
  // B: lane holds column n0+lm(+16j); same K layout as A (j=0..7 -> K=kbase+j, j=8..15 -> K=kbase+16+(j-8))
  const _Float16* brow = Wt + (size_t)(n0 + lm) * K + (lh << 3);

  for (int k0 = 0; k0 < K; k0 += 32) {
    v8h alo = *(const v8h*)(arow + k0);
    v8h ahi = *(const v8h*)(arow + k0 + 16);
    v16h a;
#pragma unroll
    for (int i = 0; i < 8; ++i) { a[i] = alo[i]; a[8 + i] = ahi[i]; }
#pragma unroll
    for (int j = 0; j < 4; ++j) {
      v8h blo = *(const v8h*)(brow + (size_t)(j << 4) * K + k0);
      v8h bhi = *(const v8h*)(brow + (size_t)(j << 4) * K + k0 + 16);
      v16h b;
#pragma unroll
      for (int i = 0; i < 8; ++i) { b[i] = blo[i]; b[8 + i] = bhi[i]; }
      acc[j] = __builtin_amdgcn_wmma_f32_16x16x32_f16(
          /*neg_a=*/false, a, /*neg_b=*/false, b,
          /*c_mod=*/(short)0, acc[j], /*reuse_a=*/false, /*reuse_b=*/false);
    }
  }

  // C/D layout: VGPR r, lanes 0-15: (M=r, N=lane); lanes 16-31: (M=r+8, N=lane-16)
#pragma unroll
  for (int j = 0; j < 4; ++j) {
    int col = n0 + (j << 4) + lm;
    float bv = bias[col];
    float* yp = Y + (size_t)(m0 + (lh << 3)) * NC + col;
#pragma unroll
    for (int r = 0; r < 8; ++r) yp[(size_t)r * NC] = acc[j][r] + bv;
  }
}

// ---------------------------------------------------------------------------
// Float atomic max via integer atomics (sign-aware)
// ---------------------------------------------------------------------------
__device__ __forceinline__ void atomicMaxF(float* addr, float val) {
  if (val >= 0.0f) {
    atomicMax((int*)addr, __float_as_int(val));
  } else {
    atomicMin((unsigned int*)addr, __float_as_uint(val));
  }
}

__device__ __forceinline__ float waveReduceSum(float p) {
#pragma unroll
  for (int off = 16; off > 0; off >>= 1) p += __shfl_xor(p, off);
  return p;
}

// ---------------------------------------------------------------------------
// conv2 edge passes (H=2).  qkvs row: [q0|q1|k0|k1|v0|v1|s(128)] (512 f32)
// ---------------------------------------------------------------------------
__global__ __launch_bounds__(256) void edge_logits_h2(const float* __restrict__ qkvs,
                                                      const int* __restrict__ src,
                                                      const int* __restrict__ dst,
                                                      float* __restrict__ alpha,
                                                      float* __restrict__ mbuf, int E) {
  int wid = (blockIdx.x * blockDim.x + threadIdx.x) >> 5;
  if (wid >= E) return;
  int lane = threadIdx.x & 31;
  int s = src[wid], d = dst[wid];
  const float* qd = qkvs + (size_t)d * 512;
  const float* ks = qkvs + (size_t)s * 512 + 128;
#pragma unroll
  for (int h = 0; h < 2; ++h) {
    float2 qv = *(const float2*)(qd + h * 64 + 2 * lane);
    float2 kv = *(const float2*)(ks + h * 64 + 2 * lane);
    float p = waveReduceSum(qv.x * kv.x + qv.y * kv.y) * 0.125f;   // 1/sqrt(64)
    if (lane == 0) {
      alpha[(size_t)wid * 2 + h] = p;
      atomicMaxF(mbuf + (size_t)d * 2 + h, p);
    }
  }
}

__global__ __launch_bounds__(256) void edge_scatter_h2(const float* __restrict__ qkvs,
                                                       const int* __restrict__ src,
                                                       const int* __restrict__ dst,
                                                       const float* __restrict__ alpha,
                                                       const float* __restrict__ mbuf,
                                                       float* __restrict__ den,
                                                       float* __restrict__ attn, int E) {
  int wid = (blockIdx.x * blockDim.x + threadIdx.x) >> 5;
  if (wid >= E) return;
  int lane = threadIdx.x & 31;
  int s = src[wid], d = dst[wid];
  const float* vs = qkvs + (size_t)s * 512 + 256;
#pragma unroll
  for (int h = 0; h < 2; ++h) {
    float w = __expf(alpha[(size_t)wid * 2 + h] - mbuf[(size_t)d * 2 + h]);
    if (lane == 0) atomicAdd(den + (size_t)d * 2 + h, w);
    float2 vv = *(const float2*)(vs + h * 64 + 2 * lane);
    float* ap = attn + (size_t)d * 128 + h * 64 + 2 * lane;
    atomicAdd(ap, w * vv.x);
    atomicAdd(ap + 1, w * vv.y);
  }
}

// h_pre = attn/den + skip; accumulate per-channel sum / sumsq for GraphNorm
__global__ __launch_bounds__(128) void finalize2_stats(const float* __restrict__ qkvs,
                                                       const float* __restrict__ attn,
                                                       const float* __restrict__ den,
                                                       float* __restrict__ hpre,
                                                       float* __restrict__ sums,
                                                       float* __restrict__ sumsq, int Nn) {
  int c = threadIdx.x;            // channel 0..127
  int h = c >> 6;
  int r0 = blockIdx.x * 128;
  float s = 0.f, sq = 0.f;
  for (int i = 0; i < 128; ++i) {
    int n = r0 + i;
    if (n >= Nn) break;
    float dn = den[(size_t)n * 2 + h] + 1e-16f;
    float v = attn[(size_t)n * 128 + c] / dn + qkvs[(size_t)n * 512 + 384 + c];
    hpre[(size_t)n * 128 + c] = v;
    s += v; sq += v * v;
  }
  atomicAdd(sums + c, s);
  atomicAdd(sumsq + c, sq);
}

// GraphNorm folded to y = A*x + B per channel
__global__ __launch_bounds__(128) void gn_coef(const float* __restrict__ sums,
                                               const float* __restrict__ sumsq,
                                               const float* __restrict__ wgt,
                                               const float* __restrict__ bias,
                                               const float* __restrict__ mscale,
                                               float* __restrict__ cA, float* __restrict__ cB,
                                               float invN) {
  int c = threadIdx.x;
  float mu = sums[c] * invN;
  float ms = mu * mscale[c];
  float var = sumsq[c] * invN - 2.f * ms * mu + ms * ms;
  float A = wgt[c] * rsqrtf(var + 1e-5f);
  cA[c] = A;
  cB[c] = bias[c] - A * ms;
}

// mish(A*h + B) -> f16 (input to conv3 GEMM)
__global__ __launch_bounds__(256) void apply_norm_mish(const float* __restrict__ hpre,
                                                       const float* __restrict__ cA,
                                                       const float* __restrict__ cB,
                                                       _Float16* __restrict__ hh, size_t total) {
  size_t t = (size_t)blockIdx.x * blockDim.x + threadIdx.x;
  if (t >= total) return;
  int c = (int)(t & 127);
  float y = cA[c] * hpre[t] + cB[c];
  float sp = (y > 20.f) ? y : log1pf(__expf(y));
  hh[t] = (_Float16)(y * tanhf(sp));
}

// ---------------------------------------------------------------------------
// conv3+conv4 fused edge passes (H=1 each). qkvsX row: [q|k|v|s] (256 f32)
// ---------------------------------------------------------------------------
__global__ __launch_bounds__(256) void edge_logits_34(const float* __restrict__ q3,
                                                      const float* __restrict__ q4,
                                                      const int* __restrict__ src,
                                                      const int* __restrict__ dst,
                                                      float* __restrict__ a3, float* __restrict__ a4,
                                                      float* __restrict__ m3, float* __restrict__ m4,
                                                      int E) {
  int wid = (blockIdx.x * blockDim.x + threadIdx.x) >> 5;
  if (wid >= E) return;
  int lane = threadIdx.x & 31;
  int s = src[wid], d = dst[wid];
  {
    float2 qv = *(const float2*)(q3 + (size_t)d * 256 + 2 * lane);
    float2 kv = *(const float2*)(q3 + (size_t)s * 256 + 64 + 2 * lane);
    float p = waveReduceSum(qv.x * kv.x + qv.y * kv.y) * 0.125f;
    if (lane == 0) { a3[wid] = p; atomicMaxF(m3 + d, p); }
  }
  {
    float2 qv = *(const float2*)(q4 + (size_t)d * 256 + 2 * lane);
    float2 kv = *(const float2*)(q4 + (size_t)s * 256 + 64 + 2 * lane);
    float p = waveReduceSum(qv.x * kv.x + qv.y * kv.y) * 0.125f;
    if (lane == 0) { a4[wid] = p; atomicMaxF(m4 + d, p); }
  }
}

__global__ __launch_bounds__(256) void edge_scatter_34(const float* __restrict__ q3,
                                                       const float* __restrict__ q4,
                                                       const int* __restrict__ src,
                                                       const int* __restrict__ dst,
                                                       const float* __restrict__ a3,
                                                       const float* __restrict__ a4,
                                                       const float* __restrict__ m3,
                                                       const float* __restrict__ m4,
                                                       float* __restrict__ den3, float* __restrict__ den4,
                                                       float* __restrict__ at3, float* __restrict__ at4,
                                                       int E) {
  int wid = (blockIdx.x * blockDim.x + threadIdx.x) >> 5;
  if (wid >= E) return;
  int lane = threadIdx.x & 31;
  int s = src[wid], d = dst[wid];
  {
    float w = __expf(a3[wid] - m3[d]);
    if (lane == 0) atomicAdd(den3 + d, w);
    float2 vv = *(const float2*)(q3 + (size_t)s * 256 + 128 + 2 * lane);
    float* ap = at3 + (size_t)d * 64 + 2 * lane;
    atomicAdd(ap, w * vv.x);
    atomicAdd(ap + 1, w * vv.y);
  }
  {
    float w = __expf(a4[wid] - m4[d]);
    if (lane == 0) atomicAdd(den4 + d, w);
    float2 vv = *(const float2*)(q4 + (size_t)s * 256 + 128 + 2 * lane);
    float* ap = at4 + (size_t)d * 64 + 2 * lane;
    atomicAdd(ap, w * vv.x);
    atomicAdd(ap + 1, w * vv.y);
  }
}

__global__ __launch_bounds__(256) void final_out(const float* __restrict__ q3,
                                                 const float* __restrict__ q4,
                                                 const float* __restrict__ at3,
                                                 const float* __restrict__ at4,
                                                 const float* __restrict__ den3,
                                                 const float* __restrict__ den4,
                                                 float* __restrict__ out, size_t total) {
  size_t t = (size_t)blockIdx.x * blockDim.x + threadIdx.x;
  if (t >= total) return;
  size_t n = t >> 6;
  int c = (int)(t & 63);
  float o = at3[t] / (den3[n] + 1e-16f) + q3[n * 256 + 192 + c]
          + at4[t] / (den4[n] + 1e-16f) + q4[n * 256 + 192 + c];
  out[t] = o;
}

// ---------------------------------------------------------------------------
// Host launch
// ---------------------------------------------------------------------------
extern "C" void kernel_launch(void* const* d_in, const int* in_sizes, int n_in,
                              void* d_out, int out_size, void* d_ws, size_t ws_size,
                              hipStream_t stream) {
  (void)in_sizes; (void)n_in; (void)out_size; (void)ws_size;
  constexpr int N = NN, E = EE;

  const float* x   = (const float*)d_in[0];
  const int*   ei  = (const int*)d_in[1];
  const int*   eSrc = ei;
  const int*   eDst = ei + E;
  // conv2 params
  const float *W2q = (const float*)d_in[2],  *b2q = (const float*)d_in[3];
  const float *W2k = (const float*)d_in[4],  *b2k = (const float*)d_in[5];
  const float *W2v = (const float*)d_in[6],  *b2v = (const float*)d_in[7];
  const float *W2s = (const float*)d_in[8],  *b2s = (const float*)d_in[9];
  // graph norm
  const float *gw = (const float*)d_in[10], *gb = (const float*)d_in[11], *gms = (const float*)d_in[12];
  // conv3 params
  const float *W3q = (const float*)d_in[13], *b3q = (const float*)d_in[14];
  const float *W3k = (const float*)d_in[15], *b3k = (const float*)d_in[16];
  const float *W3v = (const float*)d_in[17], *b3v = (const float*)d_in[18];
  const float *W3s = (const float*)d_in[19], *b3s = (const float*)d_in[20];
  // conv4 params
  const float *W4q = (const float*)d_in[21], *b4q = (const float*)d_in[22];
  const float *W4k = (const float*)d_in[23], *b4k = (const float*)d_in[24];
  const float *W4v = (const float*)d_in[25], *b4v = (const float*)d_in[26];
  const float *W4s = (const float*)d_in[27], *b4s = (const float*)d_in[28];

  // ---- workspace carve-out (regions reused across phases) ----
  char* base = (char*)d_ws;
  size_t off = 0;
  auto take = [&](size_t bytes) -> void* {
    void* p = base + off;
    off = (off + bytes + 255) & ~(size_t)255;
    return p;
  };
  float*    qkvs  = (float*)take((size_t)N * 512 * 4);  // conv2 QKVS; later conv3|conv4 QKVS
  float*    alpha = (float*)take((size_t)E * 2 * 4);    // alpha2; later alpha3|alpha4
  float*    mbuf  = (float*)take((size_t)N * 2 * 4);    // m2; later m3|m4
  float*    den   = (float*)take((size_t)N * 2 * 4);    // den2; later den3|den4
  float*    attn  = (float*)take((size_t)N * 128 * 4);  // attn2; later attn3|attn4
  float*    hpre  = (float*)take((size_t)N * 128 * 4);
  _Float16* xh    = (_Float16*)take((size_t)N * 256 * 2);
  _Float16* hh    = (_Float16*)take((size_t)N * 128 * 2);
  _Float16* W2t   = (_Float16*)take((size_t)512 * 256 * 2);
  _Float16* W3t   = (_Float16*)take((size_t)256 * 128 * 2);
  _Float16* W4t   = (_Float16*)take((size_t)256 * 256 * 2);
  float*    b2f   = (float*)take(512 * 4);
  float*    b3f   = (float*)take(256 * 4);
  float*    b4f   = (float*)take(256 * 4);
  float*    sums  = (float*)take(128 * 4);
  float*    sumsq = (float*)take(128 * 4);
  float*    cA    = (float*)take(128 * 4);
  float*    cB    = (float*)take(128 * 4);

  float* qkvs3 = qkvs;                       // [N][256]
  float* qkvs4 = qkvs + (size_t)N * 256;     // [N][256]
  float* a3 = alpha, *a4 = alpha + E;
  float* m3 = mbuf,  *m4 = mbuf + N;
  float* den3 = den, *den4 = den + N;
  float* at3 = attn, *at4 = attn + (size_t)N * 64;

  auto fill = [&](float* p, float v, size_t cnt) {
    fill_f32<<<(unsigned)((cnt + 255) / 256), 256, 0, stream>>>(p, v, cnt);
  };

  // ---- phase 0: precision prep ----
  cast_f32_to_f16<<<(unsigned)(((size_t)N * 256 + 255) / 256), 256, 0, stream>>>(x, xh, (size_t)N * 256);
  prep_w<<<(512 * 256) / 256, 256, 0, stream>>>(W2q, W2k, W2v, W2s, b2q, b2k, b2v, b2s, W2t, b2f, 256, 128);
  prep_w<<<(256 * 128) / 256, 256, 0, stream>>>(W3q, W3k, W3v, W3s, b3q, b3k, b3v, b3s, W3t, b3f, 128, 64);
  prep_w<<<(256 * 256) / 256, 256, 0, stream>>>(W4q, W4k, W4v, W4s, b4q, b4k, b4v, b4s, W4t, b4f, 256, 64);

  // ---- phase 1: conv2 ----
  fill(attn, 0.f, (size_t)N * 128);
  fill(den, 0.f, (size_t)N * 2);
  fill(mbuf, -INFINITY, (size_t)N * 2);
  fill(sums, 0.f, 128);
  fill(sumsq, 0.f, 128);

  {  // X[N,256] @ W2t -> qkvs [N,512]
    int tiles = (N / 16) * (512 / 64);
    wmma_gemm_f16<<<(tiles + 7) / 8, 256, 0, stream>>>(xh, W2t, b2f, qkvs, N, 256, 512);
  }
  edge_logits_h2<<<(E * 32) / 256, 256, 0, stream>>>(qkvs, eSrc, eDst, alpha, mbuf, E);
  edge_scatter_h2<<<(E * 32) / 256, 256, 0, stream>>>(qkvs, eSrc, eDst, alpha, mbuf, den, attn, E);
  finalize2_stats<<<(N + 127) / 128, 128, 0, stream>>>(qkvs, attn, den, hpre, sums, sumsq, N);
  gn_coef<<<1, 128, 0, stream>>>(sums, sumsq, gw, gb, gms, cA, cB, 1.0f / (float)N);
  apply_norm_mish<<<(unsigned)(((size_t)N * 128 + 255) / 256), 256, 0, stream>>>(hpre, cA, cB, hh, (size_t)N * 128);

  // ---- phase 2: conv3 (on h) + conv4 (on x) ----
  fill(attn, 0.f, (size_t)N * 128);          // attn3|attn4
  fill(den, 0.f, (size_t)N * 2);             // den3|den4
  fill(mbuf, -INFINITY, (size_t)N * 2);      // m3|m4
  {
    int tiles = (N / 16) * (256 / 64);
    wmma_gemm_f16<<<(tiles + 7) / 8, 256, 0, stream>>>(hh, W3t, b3f, qkvs3, N, 128, 256);
    wmma_gemm_f16<<<(tiles + 7) / 8, 256, 0, stream>>>(xh, W4t, b4f, qkvs4, N, 256, 256);
  }
  edge_logits_34<<<(E * 32) / 256, 256, 0, stream>>>(qkvs3, qkvs4, eSrc, eDst, a3, a4, m3, m4, E);
  edge_scatter_34<<<(E * 32) / 256, 256, 0, stream>>>(qkvs3, qkvs4, eSrc, eDst, a3, a4, m3, m4,
                                                      den3, den4, at3, at4, E);
  final_out<<<(unsigned)(((size_t)N * 64 + 255) / 256), 256, 0, stream>>>(
      qkvs3, qkvs4, at3, at4, den3, den4, (float*)d_out, (size_t)N * 64);
}
